// CausalSelfAttention_5239860101719
// MI455X (gfx1250) — matrix operand order repaired and checked by reading, not compiled, not run on hardware
//
#include <hip/hip_runtime.h>
#include <hip/hip_bf16.h>
#include <math.h>

typedef __attribute__((ext_vector_type(16))) __bf16 v16bf;
typedef __attribute__((ext_vector_type(8)))  float  v8f;
typedef __attribute__((ext_vector_type(4)))  __bf16 bf4;
typedef __attribute__((ext_vector_type(4)))  float  f4;

#define DMODEL 768
#define NHEADS 12
#define DHEAD  64
#define BATCH  2
#define SEQ    4096
#define NQKV   (3*DMODEL)       // 2304
#define MROWS  (BATCH*SEQ)      // 8192

// ---------- helpers ----------

__device__ __forceinline__ __bf16 f2bf(float f) {
    unsigned u = __float_as_uint(f);
    u += 0x7fffu + ((u >> 16) & 1u);          // round-to-nearest-even
    unsigned short s = (unsigned short)(u >> 16);
    __bf16 r;
    __builtin_memcpy(&r, &s, sizeof(r));
    return r;
}

__device__ __forceinline__ v8f bwmma(v16bf a, v16bf b, v8f c) {
    return __builtin_amdgcn_wmma_f32_16x16x32_bf16(
        /*neg_a=*/false, a, /*neg_b=*/false, b,
        /*c_mod=*/(short)0, c, /*reuse_a=*/false, /*reuse_b=*/false);
}

// A fragment: 16x32 (MxK) bf16, tile stored [M][K] row-major with stride.
// ISA layout: lanes 0-15 M=lane; elements 0..7 -> K = 8*half + e,
//             elements 8..15 -> K = 16 + 8*half + (e-8).
// Both 8-element runs are 16B-contiguous -> 2x ds_load_b128 per fragment.
__device__ __forceinline__ v16bf load_a(const __bf16* base, int stride,
                                        int row_off, int k_off, int lane) {
    const int row  = row_off + (lane & 15);
    const int half = (lane >> 4) & 1;
    const __bf16* p = base + row * stride + k_off;
    v16bf a;
#pragma unroll
    for (int e = 0; e < 8; ++e) a[e] = p[8 * half + e];
#pragma unroll
    for (int e = 0; e < 8; ++e) a[8 + e] = p[16 + 8 * half + e];
    return a;
}

// B fragment: 32x16 (KxN) bf16 where logical element [k][n] lives at
// base[n*stride + k] (fragment-major / transposed tile).
// ISA layout: lane col = lane&15; elements 0..7 -> K = 8*half + e,
//             elements 8..15 -> K = 16 + 8*half + (e-8).
// Both 8-element runs are 16B-contiguous -> 2x ds_load_b128 per fragment.
__device__ __forceinline__ v16bf load_bT(const __bf16* base, int stride,
                                         int k_off, int col_off, int lane) {
    const int col  = col_off + (lane & 15);
    const int half = (lane >> 4) & 1;
    const __bf16* p = base + col * stride + k_off;
    v16bf b;
#pragma unroll
    for (int e = 0; e < 8; ++e) b[e] = p[8 * half + e];
#pragma unroll
    for (int e = 0; e < 8; ++e) b[8 + e] = p[16 + 8 * half + e];
    return b;
}

// ---------- kernels ----------

__global__ void cast_f32_bf16(const float* __restrict__ src,
                              __bf16* __restrict__ dst, int n4) {
    int i = blockIdx.x * blockDim.x + threadIdx.x;
    const int stride = gridDim.x * blockDim.x;
    for (; i < n4; i += stride) {
        f4 v = *(const f4*)(src + 4 * (size_t)i);
        bf4 r;
#pragma unroll
        for (int j = 0; j < 4; ++j) r[j] = f2bf(v[j]);
        *(bf4*)(dst + 4 * (size_t)i) = r;
    }
}

// QKV projection: [8192x768] @ [768x2304] + bias, scatter to (3,B,H,L,64) bf16.
// Q plane pre-scaled by 1/sqrt(DHEAD).
__launch_bounds__(256)
__global__ void qkv_gemm(const __bf16* __restrict__ X,
                         const __bf16* __restrict__ W,
                         const float*  __restrict__ bias,
                         __bf16* __restrict__ QKV) {
    __shared__ __align__(16) __bf16 lA[128 * 40];   // [m][k], stride 40 (80B)
    __shared__ __align__(16) __bf16 lBt[64 * 40];   // [n][k], stride 40 (80B)

    const int t    = threadIdx.x;
    const int lane = t & 31;
    const int wave = t >> 5;
    const int wm   = wave & 3;    // 4 sub-tiles of 32 rows
    const int wn   = wave >> 2;   // 2 sub-tiles of 32 cols
    const int m0   = blockIdx.y * 128;
    const int n0   = blockIdx.x * 64;

    v8f acc[2][2] = {};
    for (int kb = 0; kb < DMODEL; kb += 32) {
        for (int i = t; i < 128 * 8; i += 256) {
            int r = i >> 3, c = (i & 7) << 2;
            const __bf16* gp = X + (size_t)(m0 + r) * DMODEL + kb + c;
            __builtin_prefetch(gp + 32, 0, 0);        // speculative, next k-tile
            *(bf4*)(lA + r * 40 + c) = *(const bf4*)gp;
        }
        for (int i = t; i < 32 * 16; i += 256) {
            int k = i >> 4, c = (i & 15) << 2;
            bf4 v = *(const bf4*)(W + (size_t)(kb + k) * NQKV + n0 + c);
#pragma unroll
            for (int j = 0; j < 4; ++j) lBt[(c + j) * 40 + k] = v[j];
        }
        __syncthreads();
        v16bf a0 = load_a (lA, 40, wm * 32,      0, lane);
        v16bf a1 = load_a (lA, 40, wm * 32 + 16, 0, lane);
        v16bf b0 = load_bT(lBt, 40, 0, wn * 32,      lane);
        v16bf b1 = load_bT(lBt, 40, 0, wn * 32 + 16, lane);
        acc[0][0] = bwmma(a0, b0, acc[0][0]);
        acc[0][1] = bwmma(a0, b1, acc[0][1]);
        acc[1][0] = bwmma(a1, b0, acc[1][0]);
        acc[1][1] = bwmma(a1, b1, acc[1][1]);
        __syncthreads();
    }

    const int half = (lane >> 4) & 1;
    const int cl   = lane & 15;
#pragma unroll
    for (int mi = 0; mi < 2; ++mi)
#pragma unroll
        for (int ni = 0; ni < 2; ++ni) {
            const int n     = n0 + wn * 32 + ni * 16 + cl;
            const int which = n / DMODEL;
            const int rem   = n % DMODEL;
            const int h     = rem >> 6;
            const int d     = rem & 63;
            const float sc  = (which == 0) ? 0.125f : 1.0f;  // 1/sqrt(64) on Q
            const float bv  = bias[n];
#pragma unroll
            for (int v = 0; v < 8; ++v) {
                const int m    = m0 + wm * 32 + mi * 16 + v + 8 * half;
                const int bidx = m >> 12;       // / SEQ
                const int lpos = m & (SEQ - 1);
                const float val = (acc[mi][ni][v] + bv) * sc;
                const size_t idx =
                    (size_t)which * ((size_t)BATCH * NHEADS * SEQ * DHEAD) +
                    (((size_t)(bidx * NHEADS + h)) * SEQ + lpos) * DHEAD + d;
                QKV[idx] = f2bf(val);
            }
        }
}

// Flash attention: one (b,h) x 64-query tile per block, 4 waves x 16 rows.
__launch_bounds__(128)
__global__ void flash_attn(const __bf16* __restrict__ QKV,
                           __bf16* __restrict__ O) {
    __shared__ __align__(16) __bf16 lQ [64 * 72];   // [q][d],   stride 72 (144B)
    __shared__ __align__(16) __bf16 lK [32 * 72];   // [key][d], stride 72
    __shared__ __align__(16) __bf16 lVt[64 * 40];   // [d][key], stride 40 (80B)
    __shared__ __align__(16) __bf16 lP [4 * 16 * 40];

    const int t    = threadIdx.x;
    const int lane = t & 31;
    const int wave = t >> 5;
    const int half = (lane >> 4) & 1;
    const int cl   = lane & 15;

    const int bh = blockIdx.y;          // b*NHEADS + h
    const int q0 = blockIdx.x * 64;
    const size_t plane = (size_t)BATCH * NHEADS * SEQ * DHEAD;
    const __bf16* Q = QKV +              (size_t)bh * SEQ * DHEAD;
    const __bf16* K = QKV + plane +      (size_t)bh * SEQ * DHEAD;
    const __bf16* V = QKV + 2 * plane +  (size_t)bh * SEQ * DHEAD;

    for (int i = t; i < 64 * 16; i += 128) {
        int r = i >> 4, c = (i & 15) << 2;
        *(bf4*)(lQ + r * 72 + c) = *(const bf4*)(Q + (size_t)(q0 + r) * DHEAD + c);
    }
    __syncthreads();

    const v16bf qa0 = load_a(lQ, 72, wave * 16,  0, lane);
    const v16bf qa1 = load_a(lQ, 72, wave * 16, 32, lane);

    float mstat[8], lstat[8];
#pragma unroll
    for (int v = 0; v < 8; ++v) { mstat[v] = -1e30f; lstat[v] = 0.0f; }
    v8f o[4] = {};

    const int qmax = q0 + wave * 16 + 15;
    const int nkb  = 2 * blockIdx.x + 2;       // causal: keys up to end of Q tile
    for (int kb = 0; kb < nkb; ++kb) {
        __syncthreads();                        // previous tile fully consumed
        const int k0 = kb * 32;
        for (int i = t; i < 32 * 16; i += 128) {
            int r = i >> 4, c = (i & 15) << 2;
            const __bf16* kp = K + (size_t)(k0 + r) * DHEAD + c;
            const __bf16* vp = V + (size_t)(k0 + r) * DHEAD + c;
            __builtin_prefetch(kp + 32 * DHEAD, 0, 0);   // speculative, next tile
            __builtin_prefetch(vp + 32 * DHEAD, 0, 0);
            *(bf4*)(lK + r * 72 + c) = *(const bf4*)kp;
            bf4 vv = *(const bf4*)vp;
#pragma unroll
            for (int j = 0; j < 4; ++j) lVt[(c + j) * 40 + r] = vv[j];
        }
        __syncthreads();

        if (k0 <= qmax) {                       // wave-uniform: EXEC stays all-1s
            v8f s0 = {}, s1 = {};
            s0 = bwmma(qa0, load_bT(lK, 72,  0,  0, lane), s0);
            s0 = bwmma(qa1, load_bT(lK, 72, 32,  0, lane), s0);
            s1 = bwmma(qa0, load_bT(lK, 72,  0, 16, lane), s1);
            s1 = bwmma(qa1, load_bT(lK, 72, 32, 16, lane), s1);

            float p0[8], p1[8], sc[8];
#pragma unroll
            for (int v = 0; v < 8; ++v) {
                const int qrow = q0 + wave * 16 + v + 8 * half;
                const float x0 = (k0 + cl      <= qrow) ? s0[v] : -1e30f;
                const float x1 = (k0 + 16 + cl <= qrow) ? s1[v] : -1e30f;
                float rmax = fmaxf(x0, x1);
                rmax = fmaxf(rmax, __shfl_xor(rmax, 1, 32));
                rmax = fmaxf(rmax, __shfl_xor(rmax, 2, 32));
                rmax = fmaxf(rmax, __shfl_xor(rmax, 4, 32));
                rmax = fmaxf(rmax, __shfl_xor(rmax, 8, 32));
                const float mnew = fmaxf(mstat[v], rmax);
                const float corr = __expf(mstat[v] - mnew);
                const float e0 = __expf(x0 - mnew);
                const float e1 = __expf(x1 - mnew);
                float rsum = e0 + e1;
                rsum += __shfl_xor(rsum, 1, 32);
                rsum += __shfl_xor(rsum, 2, 32);
                rsum += __shfl_xor(rsum, 4, 32);
                rsum += __shfl_xor(rsum, 8, 32);
                lstat[v] = lstat[v] * corr + rsum;
                mstat[v] = mnew;
                sc[v] = corr; p0[v] = e0; p1[v] = e1;
            }
#pragma unroll
            for (int j = 0; j < 4; ++j)
#pragma unroll
                for (int v = 0; v < 8; ++v) o[j][v] *= sc[v];

            // C-layout -> A-layout via per-wave LDS staging tile
            __bf16* myP = lP + wave * 16 * 40;
#pragma unroll
            for (int v = 0; v < 8; ++v) {
                const int row = v + 8 * half;
                myP[row * 40 + cl]      = f2bf(p0[v]);
                myP[row * 40 + 16 + cl] = f2bf(p1[v]);
            }
            const v16bf pa = load_a(myP, 40, 0, 0, lane);
#pragma unroll
            for (int j = 0; j < 4; ++j)
                o[j] = bwmma(pa, load_bT(lVt, 40, 0, j * 16, lane), o[j]);
        }
    }

    const int b_idx = bh / NHEADS;
    const int h     = bh % NHEADS;
#pragma unroll
    for (int v = 0; v < 8; ++v) {
        const float inv = 1.0f / lstat[v];
        const int qrow  = q0 + wave * 16 + v + 8 * half;
        const size_t rb = ((size_t)b_idx * SEQ + qrow) * DMODEL + h * DHEAD;
#pragma unroll
        for (int j = 0; j < 4; ++j)
            O[rb + j * 16 + cl] = f2bf(o[j][v] * inv);
    }
}

// Output projection: [8192x768] @ [768x768] + bias -> f32.
__launch_bounds__(256)
__global__ void out_gemm(const __bf16* __restrict__ A,
                         const __bf16* __restrict__ W,
                         const float*  __restrict__ bias,
                         float* __restrict__ out) {
    __shared__ __align__(16) __bf16 lA[128 * 40];   // [m][k]
    __shared__ __align__(16) __bf16 lBt[64 * 40];   // [n][k]

    const int t    = threadIdx.x;
    const int lane = t & 31;
    const int wave = t >> 5;
    const int wm   = wave & 3;
    const int wn   = wave >> 2;
    const int m0   = blockIdx.y * 128;
    const int n0   = blockIdx.x * 64;

    v8f acc[2][2] = {};
    for (int kb = 0; kb < DMODEL; kb += 32) {
        for (int i = t; i < 128 * 8; i += 256) {
            int r = i >> 3, c = (i & 7) << 2;
            const __bf16* gp = A + (size_t)(m0 + r) * DMODEL + kb + c;
            __builtin_prefetch(gp + 32, 0, 0);
            *(bf4*)(lA + r * 40 + c) = *(const bf4*)gp;
        }
        for (int i = t; i < 32 * 16; i += 256) {
            int k = i >> 4, c = (i & 15) << 2;
            bf4 v = *(const bf4*)(W + (size_t)(kb + k) * DMODEL + n0 + c);
#pragma unroll
            for (int j = 0; j < 4; ++j) lBt[(c + j) * 40 + k] = v[j];
        }
        __syncthreads();
        v16bf a0 = load_a (lA, 40, wm * 32,      0, lane);
        v16bf a1 = load_a (lA, 40, wm * 32 + 16, 0, lane);
        v16bf b0 = load_bT(lBt, 40, 0, wn * 32,      lane);
        v16bf b1 = load_bT(lBt, 40, 0, wn * 32 + 16, lane);
        acc[0][0] = bwmma(a0, b0, acc[0][0]);
        acc[0][1] = bwmma(a0, b1, acc[0][1]);
        acc[1][0] = bwmma(a1, b0, acc[1][0]);
        acc[1][1] = bwmma(a1, b1, acc[1][1]);
        __syncthreads();
    }

    const int half = (lane >> 4) & 1;
    const int cl   = lane & 15;
#pragma unroll
    for (int mi = 0; mi < 2; ++mi)
#pragma unroll
        for (int ni = 0; ni < 2; ++ni) {
            const int n    = n0 + wn * 32 + ni * 16 + cl;
            const float bv = bias[n];
#pragma unroll
            for (int v = 0; v < 8; ++v) {
                const int m = m0 + wm * 32 + mi * 16 + v + 8 * half;
                out[(size_t)m * DMODEL + n] = acc[mi][ni][v] + bv;
            }
        }
}

// ---------- launch ----------

extern "C" void kernel_launch(void* const* d_in, const int* in_sizes, int n_in,
                              void* d_out, int out_size, void* d_ws, size_t ws_size,
                              hipStream_t stream) {
    const float* x     = (const float*)d_in[0];
    const float* w_qkv = (const float*)d_in[1];
    const float* b_qkv = (const float*)d_in[2];
    const float* w_out = (const float*)d_in[3];
    const float* b_out = (const float*)d_in[4];
    float* out = (float*)d_out;

    char* ws = (char*)d_ws;
    __bf16* xb    = (__bf16*)(ws);                 //  8192x768  bf16 : 12,582,912 B
    __bf16* wqkvb = (__bf16*)(ws + 12582912);      //   768x2304 bf16 :  3,538,944 B
    __bf16* woutb = (__bf16*)(ws + 16121856);      //   768x768  bf16 :  1,179,648 B
    __bf16* qkv   = (__bf16*)(ws + 17301504);      // 3xBxHxLx64 bf16 : 37,748,736 B
    __bf16* attb  = (__bf16*)(ws + 55050240);      //  8192x768  bf16 : 12,582,912 B

    cast_f32_bf16<<<dim3(512), dim3(256), 0, stream>>>(x,     xb,    (MROWS * DMODEL) / 4);
    cast_f32_bf16<<<dim3(512), dim3(256), 0, stream>>>(w_qkv, wqkvb, (DMODEL * NQKV) / 4);
    cast_f32_bf16<<<dim3(512), dim3(256), 0, stream>>>(w_out, woutb, (DMODEL * DMODEL) / 4);

    qkv_gemm<<<dim3(NQKV / 64, MROWS / 128), dim3(256), 0, stream>>>(xb, wqkvb, b_qkv, qkv);
    flash_attn<<<dim3(SEQ / 64, BATCH * NHEADS), dim3(128), 0, stream>>>(qkv, attb);
    out_gemm<<<dim3(DMODEL / 64, MROWS / 128), dim3(256), 0, stream>>>(attb, woutb, b_out, out);
}
